// VIN_3272765079886
// MI455X (gfx1250) — matrix operand order repaired and checked by reading, not compiled, not run on hardware
//
#include <hip/hip_runtime.h>
#include <cstdint>
#include <cstddef>

typedef __attribute__((ext_vector_type(16))) __bf16 v16bf;
typedef __attribute__((ext_vector_type(8)))  float  v8f;

struct bfpair { uint4 lo, hi; };   // 32 bytes == v16bf

#define BATCH 128
#define HW    64
#define HW2   4096
#define PW    66          // spatially padded width/height (1px zero ring)
#define PW2   (PW * PW)   // 4356 pixels per padded image

// ---------- utility kernels ----------
__global__ void zero_bf16(__bf16* p, long long n) {
    long long i = blockIdx.x * (long long)blockDim.x + threadIdx.x;
    if (i < n) p[i] = (__bf16)0.f;
}
__global__ void zero_f32(float* p, int n) {
    int i = blockIdx.x * blockDim.x + threadIdx.x;
    if (i < n) p[i] = 0.f;
}

// zero the 1-pixel border ring of a padded NHWC buffer (260 border px / image)
__global__ void border_zero(__bf16* buf, int Cpad) {
    long long total = (long long)BATCH * 260 * Cpad;
    for (long long i = blockIdx.x * (long long)blockDim.x + threadIdx.x;
         i < total; i += (long long)gridDim.x * blockDim.x) {
        int c = (int)(i % Cpad);
        long long q = i / Cpad;
        int n = (int)(q / 260);
        int e = (int)(q % 260);
        int py, px;
        if (e < 132) { py = (e / 66) * 65; px = e % 66; }         // top/bottom rows
        else { int e2 = e - 132; py = (e2 & 63) + 1; px = (e2 >> 6) * 65; } // side cols
        buf[(((size_t)n * PW + py) * PW + px) * Cpad + c] = (__bf16)0.f;
    }
}

// X (NCHW, Cin=2) -> padded NHWC bf16, Cpad=32 (interior only; border zeroed separately)
__global__ void xcast_nhwc(const float* __restrict__ src, __bf16* __restrict__ dst) {
    long long i = blockIdx.x * (long long)blockDim.x + threadIdx.x;  // over 128*4096*32
    if (i >= (long long)BATCH * HW2 * 32) return;
    int c = (int)(i & 31);
    long long p = i >> 5;
    int x = (int)(p & 63), y = (int)((p >> 6) & 63), n = (int)(p >> 12);
    __bf16 v = (c < 2) ? (__bf16)src[((long long)n * 2 + c) * HW2 + (y << 6) + x]
                       : (__bf16)0.f;
    dst[(((size_t)n * PW + y + 1) * PW + x + 1) * 32 + c] = v;
}

// weights (Cout,Cin,3,3) f32 -> [tap][co][ci] bf16 (pre-zeroed buffer handles padding)
__global__ void wcvt_t(const float* __restrict__ src, __bf16* __restrict__ dst,
                       int Cout, int Cin, int Cpo, int Cpi) {
    int i = blockIdx.x * blockDim.x + threadIdx.x;
    if (i >= Cout * Cin * 9) return;
    int co  = i / (Cin * 9);
    int rem = i - co * (Cin * 9);
    int ci  = rem / 9;
    int t   = rem - ci * 9;
    dst[((size_t)t * Cpo + co) * Cpi + ci] = (__bf16)src[i];
}

// ---------- 3x3 conv, padded NHWC, tap-decomposed implicit GEMM, WMMA bf16 ----------
// CPI/CPO compile-time => all load addresses are one per-lane base + immediate offsets.
// Block: 256 threads = 8 waves; tile M=64 (one row), N=32 (co tile). No LDS staging,
// no predication (zero ring supplies halo zeros). Two accumulators break WMMA RAW chains.
template<int CPI, int CPO>
__global__ __launch_bounds__(256) void conv3x3_wmma(
    const __bf16* __restrict__ in,    // padded NHWC [B][66][66][CPI]
    const __bf16* __restrict__ wgt,   // [9][CPO][CPI]
    const float* __restrict__ bias,   // [Cout]
    __bf16* __restrict__ out,         // padded NHWC [B][66][66][CPO]
    float* __restrict__ stats,        // [2*Cout]
    int Cout)
{
    constexpr int KC = CPI / 32;
    __shared__ float ssum[32], ssq[32];

    const int tid   = threadIdx.x;
    const int lane  = tid & 31;
    const int wvid  = tid >> 5;
    const int subm  = wvid & 3;
    const int subn  = wvid >> 2;
    const int n     = blockIdx.y >> 6;
    const int y     = blockIdx.y & 63;          // output row (interior coords)
    const int n0    = blockIdx.x << 5;
    const int khalf = lane >> 4;
    const int l15   = lane & 15;
    const int xm    = (subm << 4) + l15;        // x position this lane owns in A
    const int co    = n0 + (subn << 4) + l15;   // output channel this lane owns in B/C

    if (tid < 32) { ssum[tid] = 0.f; ssq[tid] = 0.f; }

    // tap (ky,kx) reads padded pixel (y+ky, xm+kx): one base pointer, constant offsets
    const __bf16* abase0 = in + ((size_t)n * PW2 + (size_t)y * PW + xm) * CPI + khalf * 8;
    const __bf16* bbase0 = wgt + (size_t)co * CPI + khalf * 8;
    __builtin_prefetch(bbase0, 0, 1);           // global_prefetch_b8 for weight slab

    v8f acc0 = {}, acc1 = {};

    #pragma unroll
    for (int ky = 0; ky < 3; ++ky) {
        #pragma unroll
        for (int kx = 0; kx < 3; ++kx) {
            const __bf16* abase = abase0 + (ky * PW + kx) * CPI;            // const offs
            const __bf16* bbase = bbase0 + (size_t)(ky * 3 + kx) * CPO * CPI;
            #pragma unroll
            for (int kc = 0; kc < KC; ++kc) {
                const int k0 = kc << 5;
                uint4 alo = *(const uint4*)(abase + k0);        // K = k0+khalf*8    ..+8
                uint4 ahi = *(const uint4*)(abase + k0 + 16);   // K = k0+16+khalf*8 ..+8
                uint4 blo = *(const uint4*)(bbase + k0);        // K = k0+khalf*8    ..+8
                uint4 bhi = *(const uint4*)(bbase + k0 + 16);   // K = k0+16+khalf*8 ..+8
                v16bf af = __builtin_bit_cast(v16bf, bfpair{alo, ahi});
                v16bf bf = __builtin_bit_cast(v16bf, bfpair{blo, bhi});
                if ((((ky * 3 + kx) * KC) + kc) & 1)
                    acc1 = __builtin_amdgcn_wmma_f32_16x16x32_bf16(
                               false, af, false, bf, (short)0, acc1, false, false);
                else
                    acc0 = __builtin_amdgcn_wmma_f32_16x16x32_bf16(
                               false, af, false, bf, (short)0, acc0, false, false);
            }
        }
    }
    #pragma unroll
    for (int r = 0; r < 8; ++r) acc0[r] += acc1[r];

    // ---- write padded NHWC output + BN stats ----
    const int xbase = (subm << 4) + (khalf << 3);        // C/D: M = reg + 8*(lane>=16)
    const float bsv = (co < Cout) ? bias[co] : 0.f;      // pad channels: acc==0, bias 0
    float lsum = 0.f, lsq = 0.f;
    const size_t obase = ((size_t)n * PW2 + (size_t)(y + 1) * PW + 1) * CPO + co;
    #pragma unroll
    for (int r = 0; r < 8; ++r) {
        float valf = acc0[r] + bsv;
        out[obase + (size_t)(xbase + r) * CPO] = (__bf16)valf;
        lsum += valf;
        lsq  += valf * valf;
    }
    if (co >= Cout) { lsum = 0.f; lsq = 0.f; }
    atomicAdd(&ssum[(subn << 4) + l15], lsum);
    atomicAdd(&ssq[(subn << 4) + l15],  lsq);
    __syncthreads();
    if (tid < 32) {
        int c2 = n0 + tid;
        if (c2 < Cout) {
            atomicAdd(&stats[c2],        ssum[tid]);
            atomicAdd(&stats[Cout + c2], ssq[tid]);
        }
    }
}

// ---------- BatchNorm (training stats) + ReLU, in place, padded NHWC interior ----------
__global__ void bn_relu(__bf16* data, const float* __restrict__ stats,
                        const float* __restrict__ g, const float* __restrict__ bb,
                        int C, int Cpad)
{
    const float invn = 1.f / (float)(BATCH * HW2);
    const long long total = (long long)BATCH * HW2 * Cpad;
    for (long long i = blockIdx.x * (long long)blockDim.x + threadIdx.x;
         i < total; i += (long long)gridDim.x * blockDim.x) {
        int c = (int)(i % Cpad);
        long long p = i / Cpad;
        int x = (int)(p & 63), yy = (int)((p >> 6) & 63), n = (int)(p >> 12);
        size_t a = (((size_t)n * PW + yy + 1) * PW + x + 1) * Cpad + c;
        float res = 0.f;
        if (c < C) {
            float mean  = stats[c] * invn;
            float var   = stats[C + c] * invn - mean * mean;
            float scale = rsqrtf(var + 1e-5f) * g[c];
            res = fmaxf(((float)data[a] - mean) * scale + bb[c], 0.f);
        }
        data[a] = (__bf16)res;
    }
}

// ---------- 1x1 conv 100ch -> 1ch (reward map), padded NHWC input ----------
__global__ void rconv1x1(const __bf16* __restrict__ h, const float* __restrict__ rw,
                         float* __restrict__ r)
{
    int i = blockIdx.x * 256 + threadIdx.x;
    if (i >= BATCH * HW2) return;
    int x = i & 63, y = (i >> 6) & 63, n = i >> 12;
    const __bf16* hp = h + (((size_t)n * PW + y + 1) * PW + x + 1) * 128;
    float acc = 0.f;
    #pragma unroll 4
    for (int c = 0; c < 100; ++c) acc += (float)hp[c] * rw[c];
    r[i] = acc;
}

// ---------- qr = conv5x5(r, q_w); v0 = max_a qr ----------
__global__ __launch_bounds__(256) void qr_init(const float* __restrict__ r,
                                               const float* __restrict__ qw,
                                               float* __restrict__ qr,
                                               float* __restrict__ v0)
{
    __shared__ float sm[20][20];
    int b   = blockIdx.y;
    int tx0 = (blockIdx.x & 3) << 4;
    int ty0 = (blockIdx.x >> 2) << 4;
    int t   = threadIdx.y * 16 + threadIdx.x;
    for (int e = t; e < 400; e += 256) {
        int ly = e / 20, lx = e - ly * 20;
        int gy = ty0 + ly - 2, gx = tx0 + lx - 2;
        sm[ly][lx] = ((unsigned)gy < 64u && (unsigned)gx < 64u)
                         ? r[((size_t)b << 12) + (gy << 6) + gx] : 0.f;
    }
    __syncthreads();
    int y = ty0 + threadIdx.y, x = tx0 + threadIdx.x;
    float vmax = -3.4e38f;
    for (int a = 0; a < 10; ++a) {
        float acc = 0.f;
        #pragma unroll
        for (int ky = 0; ky < 5; ++ky)
            #pragma unroll
            for (int kx = 0; kx < 5; ++kx)
                acc += sm[threadIdx.y + ky][threadIdx.x + kx] * qw[a * 25 + ky * 5 + kx];
        qr[(((size_t)b * 10 + a) << 12) + (y << 6) + x] = acc;
        vmax = fmaxf(vmax, acc);
    }
    v0[((size_t)b << 12) + (y << 6) + x] = vmax;
}

// ---------- one VI step: v_new = max_a( qr[a] + conv5x5(v_old, w[a]) ) ----------
__global__ __launch_bounds__(256) void vi_step(const float* __restrict__ vold,
                                               const float* __restrict__ qr,
                                               const float* __restrict__ wf,
                                               float* __restrict__ vnew)
{
    __shared__ float sm[20][20];
    int b   = blockIdx.y;
    int tx0 = (blockIdx.x & 3) << 4;
    int ty0 = (blockIdx.x >> 2) << 4;
    int t   = threadIdx.y * 16 + threadIdx.x;
    for (int e = t; e < 400; e += 256) {
        int ly = e / 20, lx = e - ly * 20;
        int gy = ty0 + ly - 2, gx = tx0 + lx - 2;
        sm[ly][lx] = ((unsigned)gy < 64u && (unsigned)gx < 64u)
                         ? vold[((size_t)b << 12) + (gy << 6) + gx] : 0.f;
    }
    __syncthreads();
    int y = ty0 + threadIdx.y, x = tx0 + threadIdx.x;
    float vmax = -3.4e38f;
    for (int a = 0; a < 10; ++a) {
        float acc = qr[(((size_t)b * 10 + a) << 12) + (y << 6) + x];
        #pragma unroll
        for (int ky = 0; ky < 5; ++ky)
            #pragma unroll
            for (int kx = 0; kx < 5; ++kx)
                acc += sm[threadIdx.y + ky][threadIdx.x + kx] * wf[a * 25 + ky * 5 + kx];
        vmax = fmaxf(vmax, acc);
    }
    vnew[((size_t)b << 12) + (y << 6) + x] = vmax;
}

// ---------- critic ----------
__global__ __launch_bounds__(256) void critic_k(const float* __restrict__ v,
                                                const float* __restrict__ cw,
                                                const float* __restrict__ cb,
                                                float* __restrict__ out)
{
    __shared__ float red[256];
    int b = blockIdx.x;
    float acc = 0.f;
    for (int p = threadIdx.x; p < 4096; p += 256)
        acc += v[((size_t)b << 12) + p] * cw[p];
    red[threadIdx.x] = acc;
    __syncthreads();
    for (int s = 128; s > 0; s >>= 1) {
        if (threadIdx.x < s) red[threadIdx.x] += red[threadIdx.x + s];
        __syncthreads();
    }
    if (threadIdx.x == 0) out[b] = red[0] + cb[0];
}

// ---------- action head ----------
__global__ __launch_bounds__(128) void action_k(const float* __restrict__ obs,
                                                const float* __restrict__ w1,
                                                const float* __restrict__ b1,
                                                const float* __restrict__ w2,
                                                const float* __restrict__ b2,
                                                float* __restrict__ out)
{
    __shared__ float hid[100];
    int b = blockIdx.x;
    int t = threadIdx.x;
    if (t < 100) {
        float s = b1[t];
        #pragma unroll
        for (int k = 0; k < 24; ++k) s += obs[b * 24 + k] * w1[t * 24 + k];
        hid[t] = fmaxf(s, 0.f);
    }
    __syncthreads();
    if (t < 10) {
        float s = b2[t];
        for (int k = 0; k < 100; ++k) s += hid[k] * w2[t * 100 + k];
        out[b * 10 + t] = fmaxf(s, 0.f);
    }
}

extern "C" void kernel_launch(void* const* d_in, const int* in_sizes, int n_in,
                              void* d_out, int out_size, void* d_ws, size_t ws_size,
                              hipStream_t stream) {
    const float* X     = (const float*)d_in[0];
    const float* obs   = (const float*)d_in[1];
    const float* h1_w  = (const float*)d_in[2];
    const float* h1_b  = (const float*)d_in[3];
    const float* g1    = (const float*)d_in[4];
    const float* b1    = (const float*)d_in[5];
    const float* h2_w  = (const float*)d_in[6];
    const float* h2_b  = (const float*)d_in[7];
    const float* g2    = (const float*)d_in[8];
    const float* b2    = (const float*)d_in[9];
    const float* h3_w  = (const float*)d_in[10];
    const float* h3_b  = (const float*)d_in[11];
    const float* g3    = (const float*)d_in[12];
    const float* b3    = (const float*)d_in[13];
    const float* r_w   = (const float*)d_in[14];
    const float* q_w   = (const float*)d_in[15];
    const float* w_    = (const float*)d_in[16];
    const float* fc1_w = (const float*)d_in[17];
    const float* fc1_b = (const float*)d_in[18];
    const float* fc2_w = (const float*)d_in[19];
    const float* fc2_b = (const float*)d_in[20];
    const float* cv_w  = (const float*)d_in[21];
    const float* cv_b  = (const float*)d_in[22];
    // d_in[23] = K (device scalar); setup fixes K = 36, compiled into launch count.
    float* out = (float*)d_out;

    char* ws = (char*)d_ws;
    size_t off = 0;
    auto alloc = [&](size_t bytes) -> char* {
        char* p = ws + off;
        off += (bytes + 255) & ~(size_t)255;
        return p;
    };
    // padded NHWC activation buffers (bf16)
    __bf16* xb = (__bf16*)alloc((size_t)BATCH * PW2 * 32  * 2);  // Cpad 32
    __bf16* hA = (__bf16*)alloc((size_t)BATCH * PW2 * 160 * 2);  // h1 (160) / h3 (128)
    __bf16* hB = (__bf16*)alloc((size_t)BATCH * PW2 * 128 * 2);  // h2 (128)
    // transposed weights [tap][co_pad][ci_pad] bf16
    const long long nw1 = (long long)9 * 160 * 32;
    const long long nw2 = (long long)9 * 128 * 160;
    const long long nw3 = (long long)9 * 128 * 128;
    __bf16* w1t = (__bf16*)alloc(nw1 * 2);
    __bf16* w2t = (__bf16*)alloc(nw2 * 2);
    __bf16* w3t = (__bf16*)alloc(nw3 * 2);
    float*  stats = (float*)alloc(304 * 4);
    float*  rb  = (float*)alloc((size_t)BATCH * HW2 * 4);
    float*  qrb = (float*)alloc((size_t)BATCH * 10 * HW2 * 4);
    float*  vb0 = (float*)alloc((size_t)BATCH * HW2 * 4);
    float*  vb1 = (float*)alloc((size_t)BATCH * HW2 * 4);

    // ---- input + weight preparation ----
    const long long nXp = (long long)BATCH * HW2 * 32;
    border_zero<<<2048, 256, 0, stream>>>(xb, 32);
    xcast_nhwc<<<(unsigned)((nXp + 255) / 256), 256, 0, stream>>>(X, xb);
    zero_bf16<<<(unsigned)((nw1 + 255) / 256), 256, 0, stream>>>(w1t, nw1);
    zero_bf16<<<(unsigned)((nw2 + 255) / 256), 256, 0, stream>>>(w2t, nw2);
    zero_bf16<<<(unsigned)((nw3 + 255) / 256), 256, 0, stream>>>(w3t, nw3);
    wcvt_t<<<(150 * 2 * 9 + 255) / 256,   256, 0, stream>>>(h1_w, w1t, 150, 2,   160, 32);
    wcvt_t<<<(120 * 150 * 9 + 255) / 256, 256, 0, stream>>>(h2_w, w2t, 120, 150, 128, 160);
    wcvt_t<<<(100 * 120 * 9 + 255) / 256, 256, 0, stream>>>(h3_w, w3t, 100, 120, 128, 128);

    // ---- conv1: 2 -> 150 (CPI 32, CPO 160) ----
    border_zero<<<2048, 256, 0, stream>>>(hA, 160);
    zero_f32<<<2, 256, 0, stream>>>(stats, 300);
    conv3x3_wmma<32, 160><<<dim3(5, BATCH * HW), 256, 0, stream>>>(xb, w1t, h1_b, hA, stats, 150);
    bn_relu<<<4096, 256, 0, stream>>>(hA, stats, g1, b1, 150, 160);
    // ---- conv2: 150 -> 120 (CPI 160, CPO 128) ----
    border_zero<<<2048, 256, 0, stream>>>(hB, 128);
    zero_f32<<<2, 256, 0, stream>>>(stats, 300);
    conv3x3_wmma<160, 128><<<dim3(4, BATCH * HW), 256, 0, stream>>>(hA, w2t, h2_b, hB, stats, 120);
    bn_relu<<<4096, 256, 0, stream>>>(hB, stats, g2, b2, 120, 128);
    // ---- conv3: 120 -> 100 (CPI 128, CPO 128), reuse hA with new layout ----
    border_zero<<<2048, 256, 0, stream>>>(hA, 128);   // after conv2 finished reading hA@160
    zero_f32<<<2, 256, 0, stream>>>(stats, 300);
    conv3x3_wmma<128, 128><<<dim3(4, BATCH * HW), 256, 0, stream>>>(hB, w3t, h3_b, hA, stats, 100);
    bn_relu<<<4096, 256, 0, stream>>>(hA, stats, g3, b3, 100, 128);

    // ---- reward map + Q init + value iteration ----
    rconv1x1<<<(BATCH * HW2 + 255) / 256, 256, 0, stream>>>(hA, r_w, rb);
    qr_init<<<dim3(16, BATCH), dim3(16, 16), 0, stream>>>(rb, q_w, qrb, vb0);

    float* cur = vb0;
    float* nxt = vb1;
    for (int i = 0; i < 36; ++i) {
        vi_step<<<dim3(16, BATCH), dim3(16, 16), 0, stream>>>(cur, qrb, w_, nxt);
        float* tmp = cur; cur = nxt; nxt = tmp;
    }

    critic_k<<<BATCH, 256, 0, stream>>>(cur, cv_w, cv_b, out);
    action_k<<<BATCH, 128, 0, stream>>>(obs, fc1_w, fc1_b, fc2_w, fc2_b, out + 128);
}
